// VQKD_31336081392007
// MI455X (gfx1250) — matrix operand order repaired and checked by reading, not compiled, not run on hardware
//
#include <hip/hip_runtime.h>
#include <hip/hip_bf16.h>

// ---------------------------------------------------------------------------
// Fused VQ-KD encode + vector-quantize for MI455X (gfx1250, wave32, WMMA).
//   out = tokens (B*196 f32) | z_q (B*32*196 f32) | loss (1 f32)
// ---------------------------------------------------------------------------

typedef __attribute__((ext_vector_type(16))) __bf16 v16bf;
typedef __attribute__((ext_vector_type(2)))  __bf16 v2bf;
typedef __attribute__((ext_vector_type(8)))  float  v8f;

union Frag16 { float4 f4[2]; unsigned short u[16]; v16bf v; };

__device__ __forceinline__ unsigned short f2bf(float f) {
    unsigned int u = __float_as_uint(f);
    u += 0x7FFFu + ((u >> 16) & 1u);
    return (unsigned short)(u >> 16);
}

// packed f32x2 -> bf16x2 (hardware v_cvt_pk_bf16_f32 when available)
__device__ __forceinline__ unsigned int pk_bf16(float lo, float hi) {
#if __has_builtin(__builtin_amdgcn_cvt_pk_bf16_f32)
    union { v2bf v; unsigned int u; } cv;
    cv.v = __builtin_amdgcn_cvt_pk_bf16_f32(lo, hi);
    return cv.u;
#else
    return (unsigned int)f2bf(lo) | ((unsigned int)f2bf(hi) << 16);
#endif
}

__device__ __forceinline__ float fast_tanh(float x) {
#if __has_builtin(__builtin_amdgcn_tanhf)
    return __builtin_amdgcn_tanhf(x);
#elif __has_builtin(__builtin_amdgcn_tanh_f32)
    return __builtin_amdgcn_tanh_f32(x);
#else
    // branch-free: tanh(x) = sign(x) * (1 - t) / (1 + t), t = 2^(-2|x|*log2e)
    float ax = __builtin_fabsf(x);
    float t  = __builtin_amdgcn_exp2f(ax * -2.885390082f); // -2*log2(e)
    float r  = (1.0f - t) * __builtin_amdgcn_rcpf(1.0f + t);
    return __builtin_copysignf(r, x);
#endif
}

// ---------------------------------------------------------------------------
// Kernel 1: emb = l2norm(codebook) -> fp32 + bf16 copies; zero loss slot.
// ---------------------------------------------------------------------------
__global__ __launch_bounds__(256) void vqkd_prep(
    const float* __restrict__ cb, float* __restrict__ embf,
    unsigned int* __restrict__ embbf2, float* __restrict__ loss)
{
    int gid = blockIdx.x * blockDim.x + threadIdx.x;
    if (gid == 0) *loss = 0.0f;
    if (gid < 8192) {
        float v[32];
        float s = 0.0f;
        #pragma unroll
        for (int c = 0; c < 32; ++c) { v[c] = cb[gid * 32 + c]; s += v[c] * v[c]; }
        float inv = rsqrtf(s + 1e-12f);
        #pragma unroll
        for (int c = 0; c < 32; c += 2) {
            float e0 = v[c] * inv, e1 = v[c + 1] * inv;
            embf[gid * 32 + c]     = e0;
            embf[gid * 32 + c + 1] = e1;
            embbf2[gid * 16 + c / 2] = pk_bf16(e0, e1);
        }
    }
}

// ---------------------------------------------------------------------------
// Kernel 2: convert w1 (768x768) and w2 (32x768) to bf16 in workspace.
// ---------------------------------------------------------------------------
__global__ __launch_bounds__(256) void vqkd_cvtw(
    const float* __restrict__ W1, const float* __restrict__ W2,
    unsigned int* __restrict__ W1B2, unsigned int* __restrict__ W2B2)
{
    int p = blockIdx.x * blockDim.x + threadIdx.x;   // pair index
    if (p < 294912) {
        float2 v = ((const float2*)W1)[p];
        W1B2[p] = pk_bf16(v.x, v.y);
    } else {
        p -= 294912;
        float2 v = ((const float2*)W2)[p];
        W2B2[p] = pk_bf16(v.x, v.y);
    }
}

// ---------------------------------------------------------------------------
// Kernel 3: fully fused per-64-row tile.
// grid = 196 blocks (12544 rows / 64), block = 256 threads (8 waves).
// ---------------------------------------------------------------------------
__global__ __launch_bounds__(256) void vqkd_fused(
    const float* __restrict__ X,
    const unsigned short* __restrict__ W1B,
    const float* __restrict__ B1v,
    const unsigned short* __restrict__ W2B,
    const float* __restrict__ B2v,
    const float* __restrict__ EMBF,
    const unsigned short* __restrict__ EMBBF,
    float* __restrict__ tok, float* __restrict__ zq, float* __restrict__ loss)
{
    // ~131 KB static LDS (gfx1250 allows up to 320 KB / workgroup)
    __shared__ unsigned short Xs[64 * 776];   // 64x768 bf16 X tile (+8 pad)
    __shared__ unsigned short Hs[64 * 136];   // 64x128 bf16 h-chunk (+8 pad)
    __shared__ float          H2f[64 * 33];   // 64x32 fp32 (h2 then z)
    __shared__ unsigned short Zbf[64 * 40];   // 64x32 bf16 z (+8 pad)
    __shared__ float          RedV[64 * 2];
    __shared__ int            RedN[64 * 2];

    const int tid    = threadIdx.x;
    const int lane   = tid & 31;
    const int wave   = tid >> 5;           // 0..7
    const int lane15 = lane & 15;
    const int hf     = lane >> 4;          // wave half (0/1)
    const int row0   = blockIdx.x * 64;

    // ---- stage X tile as bf16 in LDS ----
    // each wave: 8 rows; each lane: 6 float4 (=24 floats -> 12 bf16x2) per row
    {
        #pragma unroll
        for (int rr = 0; rr < 8; ++rr) {
            const int row = wave * 8 + rr;
            const float4* src = (const float4*)(X + ((size_t)(row0 + row)) * 768);
            #pragma unroll
            for (int p = 0; p < 6; ++p) {
                const int q4 = lane + p * 32;          // float4 index in row
                float4 v = src[q4];
                uint2 pk;
                pk.x = pk_bf16(v.x, v.y);
                pk.y = pk_bf16(v.z, v.w);
                *(uint2*)&Xs[row * 776 + q4 * 4] = pk;
            }
        }
    }
    __syncthreads();

    // wave tiling: 4 row-tiles x 2 col-groups
    const int m0  = (wave >> 1) * 16;
    const int wc  = wave & 1;
    const int n0b = wc * 16;               // GEMM2 col tile (0/16)

    v8f acc2;
    #pragma unroll
    for (int r = 0; r < 8; ++r) acc2[r] = 0.0f;

    for (int kc = 0; kc < 6; ++kc) {       // 6 chunks of 128 h-columns
        // ---- GEMM1 chunk: tanh(X @ W1^T + b1) -> Hs ----
        v8f acc[4];
        #pragma unroll
        for (int jt = 0; jt < 4; ++jt)
            #pragma unroll
            for (int r = 0; r < 8; ++r) acc[jt][r] = 0.0f;

        for (int kk = 0; kk < 24; ++kk) {
            const int kbase = kk * 32;
            // A: two contiguous 8-elem bf16 runs per lane (ISA 16-bit A layout)
            Frag16 a;
            const unsigned short* xs =
                &Xs[(m0 + lane15) * 776 + kbase + hf * 8];
            a.f4[0] = *(const float4*)(xs);
            a.f4[1] = *(const float4*)(xs + 16);
            #pragma unroll
            for (int jt = 0; jt < 4; ++jt) {
                const int dl = wc * 16 + jt * 32 + lane15;
                const int dg = kc * 128 + dl;
                const unsigned short* wp =
                    W1B + (size_t)dg * 768 + kbase + hf * 8;
                Frag16 b;
                b.f4[0] = ((const float4*)wp)[0];
                b.f4[1] = ((const float4*)wp)[2];
                acc[jt] = __builtin_amdgcn_wmma_f32_16x16x32_bf16(
                    false, a.v, false, b.v, (short)0, acc[jt], false, false);
            }
        }
        #pragma unroll
        for (int jt = 0; jt < 4; ++jt) {
            const int dl = wc * 16 + jt * 32 + lane15;
            const float bias = B1v[kc * 128 + dl];
            #pragma unroll
            for (int r = 0; r < 8; ++r) {
                const int m = m0 + r + 8 * hf;      // C layout: M = r + 8*half
                Hs[m * 136 + dl] = f2bf(fast_tanh(acc[jt][r] + bias));
            }
        }
        __syncthreads();

        // ---- GEMM2 partial: h2 += Hchunk @ W2chunk^T ----
        for (int ks = 0; ks < 4; ++ks) {
            const int kbl = ks * 32;
            Frag16 a2, b2;
            const unsigned short* hp =
                &Hs[(m0 + lane15) * 136 + kbl + hf * 8];
            a2.f4[0] = *(const float4*)(hp);
            a2.f4[1] = *(const float4*)(hp + 16);
            const unsigned short* w2p =
                W2B + (size_t)(n0b + lane15) * 768 + kc * 128 + kbl + hf * 8;
            b2.f4[0] = ((const float4*)w2p)[0];
            b2.f4[1] = ((const float4*)w2p)[2];
            acc2 = __builtin_amdgcn_wmma_f32_16x16x32_bf16(
                false, a2.v, false, b2.v, (short)0, acc2, false, false);
        }
        __syncthreads();   // before next chunk overwrites Hs
    }

    // ---- h2 + b2 -> LDS fp32 ----
    {
        const int c = n0b + lane15;
        const float bias = B2v[c];
        #pragma unroll
        for (int r = 0; r < 8; ++r)
            H2f[(m0 + r + 8 * hf) * 33 + c] = acc2[r] + bias;
    }
    __syncthreads();

    // ---- row L2 normalize ----
    if (tid < 64) {
        float s = 0.0f;
        #pragma unroll
        for (int c = 0; c < 32; ++c) { float v = H2f[tid * 33 + c]; s += v * v; }
        const float inv = rsqrtf(s + 1e-12f);
        #pragma unroll
        for (int c = 0; c < 32; c += 2) {
            float z0 = H2f[tid * 33 + c] * inv;
            float z1 = H2f[tid * 33 + c + 1] * inv;
            H2f[tid * 33 + c]     = z0;
            H2f[tid * 33 + c + 1] = z1;
            *(unsigned int*)&Zbf[tid * 40 + c] = pk_bf16(z0, z1);
        }
    }
    __syncthreads();

    // ---- nearest codebook entry: argmax of z.e via WMMA (K=32) ----
    {
        const int rt = wave & 3;
        const int m0d = rt * 16;
        const int cg = wave >> 2;
        const int colbase = cg * 4096;

        Frag16 a;
        const unsigned short* zp = &Zbf[(m0d + lane15) * 40 + hf * 8];
        a.f4[0] = *(const float4*)(zp);
        a.f4[1] = *(const float4*)(zp + 16);

        float bestv[8]; int bestn[8];
        #pragma unroll
        for (int r = 0; r < 8; ++r) { bestv[r] = -3.0e38f; bestn[r] = 0; }

        #pragma unroll 2
        for (int ct = 0; ct < 256; ++ct) {
            const int e = colbase + ct * 16 + lane15;
            Frag16 b;
            const float4* p = (const float4*)(EMBBF + (size_t)e * 32 + hf * 8);
            b.f4[0] = p[0];
            b.f4[1] = p[2];
            __builtin_prefetch(EMBBF + (size_t)((e + 64) & 8191) * 32, 0, 1);
            v8f c0;
            #pragma unroll
            for (int r = 0; r < 8; ++r) c0[r] = 0.0f;
            v8f d = __builtin_amdgcn_wmma_f32_16x16x32_bf16(
                false, a.v, false, b.v, (short)0, c0, false, false);
            #pragma unroll
            for (int r = 0; r < 8; ++r)
                if (d[r] > bestv[r]) { bestv[r] = d[r]; bestn[r] = e; }
        }
        #pragma unroll
        for (int r = 0; r < 8; ++r) {
            #pragma unroll
            for (int off = 8; off >= 1; off >>= 1) {
                float vv = __shfl_xor(bestv[r], off);
                int   nn = __shfl_xor(bestn[r], off);
                if (vv > bestv[r] || (vv == bestv[r] && nn < bestn[r])) {
                    bestv[r] = vv; bestn[r] = nn;
                }
            }
            if (lane15 == 0) {
                const int m = m0d + r + 8 * hf;
                RedV[m * 2 + cg] = bestv[r];
                RedN[m * 2 + cg] = bestn[r];
            }
        }
    }
    __syncthreads();

    // ---- emit token, z_q, loss partial ----
    if (tid < 64) {
        const float v0 = RedV[tid * 2 + 0], v1 = RedV[tid * 2 + 1];
        const int   i0 = RedN[tid * 2 + 0], i1 = RedN[tid * 2 + 1];
        const int idx = (v1 > v0 || (v1 == v0 && i1 < i0)) ? i1 : i0;
        const int grow = row0 + tid;
        const int b = grow / 196;
        const int n = grow % 196;
        tok[grow] = (float)idx;
        float ls = 0.0f;
        #pragma unroll
        for (int c = 0; c < 32; ++c) {
            const float q = EMBF[idx * 32 + c];
            const float z = H2f[tid * 33 + c];
            const float d = q - z;
            ls += d * d;
            zq[((size_t)b * 32 + c) * 196 + n] = q;
        }
        atomicAdd(loss, ls * (1.0f / (12544.0f * 32.0f)));
    }
}

// ---------------------------------------------------------------------------
extern "C" void kernel_launch(void* const* d_in, const int* in_sizes, int n_in,
                              void* d_out, int out_size, void* d_ws, size_t ws_size,
                              hipStream_t stream) {
    const float* features = (const float*)d_in[0]; // raw (12544,768)
    const float* w1       = (const float*)d_in[1]; // (768,768)
    const float* b1       = (const float*)d_in[2]; // (768)
    const float* w2       = (const float*)d_in[3]; // (32,768)
    const float* b2       = (const float*)d_in[4]; // (32)
    const float* codebook = (const float*)d_in[5]; // (8192,32)

    char* ws = (char*)d_ws;
    float*          embf  = (float*)ws;           ws += 8192 * 32 * 4; // 1 MB
    unsigned short* embbf = (unsigned short*)ws;  ws += 8192 * 32 * 2; // 512 KB
    unsigned short* w1bf  = (unsigned short*)ws;  ws += 768 * 768 * 2; // 1.125 MB
    unsigned short* w2bf  = (unsigned short*)ws;  ws += 32 * 768 * 2;  // 48 KB

    float* out  = (float*)d_out;
    float* tokp = out;                    // 12544
    float* zqp  = out + 12544;            // 401408
    float* lsp  = out + 12544 + 401408;   // 1

    vqkd_prep<<<32, 256, 0, stream>>>(codebook, embf, (unsigned int*)embbf, lsp);
    vqkd_cvtw<<<1200, 256, 0, stream>>>(w1, w2, (unsigned int*)w1bf,
                                        (unsigned int*)w2bf);
    vqkd_fused<<<196, 256, 0, stream>>>(features, w1bf, b1, w2bf, b2,
                                        embf, embbf, tokp, zqp, lsp);
}
